// SurfaceAbstraction_14259291422778
// MI455X (gfx1250) — matrix operand order repaired and checked by reading, not compiled, not run on hardware
//
#include <hip/hip_runtime.h>

typedef __attribute__((ext_vector_type(16))) _Float16 v16h;
typedef __attribute__((ext_vector_type(8)))  _Float16 v8h;
typedef __attribute__((ext_vector_type(8)))  float    v8f;
typedef unsigned int u32x4 __attribute__((ext_vector_type(4)));
typedef int          i32x4 __attribute__((ext_vector_type(4)));
typedef int          i32x8 __attribute__((ext_vector_type(8)));

#define NPTS   32768
#define NSAMP  32
#define MROWS  (NPTS * NSAMP)        // 1048576 rows through the MLP
#define ROWS   128                   // rows per block (8 waves x 16)
#define KP0    104                   // 96 (padded K for 73) + 8 bank pad
#define KP1    72                    // 64 + 8 bank pad

// ---------- TDM: 1-D contiguous f16 tile, global -> LDS ----------
// D# per CDNA5 ISA ch.8: group0 {count=1, lds_addr, global_addr, type=2},
// group1 {data_size=1(2B), tensor_dim0=tile_dim0=nElems, tensor_dim1=1},
// groups 2/3 zero (<=2D tensor). One issue per block (TDM ignores EXEC).
__device__ __forceinline__ void tdm_load_1d_f16(void* ldsDst, const _Float16* gsrc, int nElems) {
    unsigned long long ga = (unsigned long long)(uintptr_t)gsrc;
    unsigned int ldsAddr  = (unsigned int)(uintptr_t)ldsDst;
    u32x4 g0 = { 1u,                                  // count=1, user mode
                 ldsAddr,                             // lds_addr
                 (unsigned int)(ga & 0xFFFFFFFFu),    // global_addr[31:0]
                 (unsigned int)((ga >> 32) & 0x1FFFFFFu) | (2u << 30) }; // [56:32] | type=2
    i32x8 g1 = { (int)(1u << 16),                     // wg_mask=0, data_size=1 (2 bytes)
                 (int)((nElems & 0xFFFF) << 16),      // tensor_dim0[15:0]
                 (int)((unsigned)nElems >> 16) | (1 << 16), // tensor_dim0[31:16] | tensor_dim1=1
                 (int)(nElems << 16),                 // tile_dim0
                 0,                                   // tile_dim1/2 = 0 (unused)
                 nElems,                              // tensor_dim0_stride[31:0]
                 0, 0 };
    i32x4 z4 = { 0, 0, 0, 0 };
#if __clang_major__ >= 23
    i32x8 z8 = { 0, 0, 0, 0, 0, 0, 0, 0 };
    __builtin_amdgcn_tensor_load_to_lds(g0, g1, z4, z4, z8, 0);
#else
    __builtin_amdgcn_tensor_load_to_lds(g0, g1, z4, z4, 0);
#endif
}

// ---------- WMMA fragment loaders (per CDNA5 ISA VGPR layouts) ----------
__device__ __forceinline__ v16h frag_a(const _Float16* As, int kp, int r0, int kb, int lane) {
    int l = lane & 15, hi = (lane >> 4) & 1;
    const _Float16* p = As + (r0 + l) * kp + kb + hi * 8;
    v8h x0 = *(const v8h*)p;
    v8h x1 = *(const v8h*)(p + 16);
    return __builtin_shufflevector(x0, x1, 0,1,2,3,4,5,6,7,8,9,10,11,12,13,14,15);
}
__device__ __forceinline__ v16h frag_b(const _Float16* Ws, int kp, int n0, int kb, int lane) {
    int l = lane & 15, hi = (lane >> 4) & 1;
    const _Float16* p = Ws + (n0 + l) * kp + kb + hi * 8;
    v8h x0 = *(const v8h*)p;
    v8h x1 = *(const v8h*)(p + 16);
    return __builtin_shufflevector(x0, x1, 0,1,2,3,4,5,6,7,8,9,10,11,12,13,14,15);
}

// C/D layout: VGPR j <-> (M = j + (lane>=16 ? 8 : 0), N = lane&15).
template<int COUT>
__device__ __forceinline__ void epilogue(v8f* acc, const float* __restrict__ bias,
                                         _Float16* __restrict__ yOut,
                                         float* ldsSum, float* ldsSq,
                                         int rowbase, int wv, int lane) {
    int l = lane & 15, hi = (lane >> 4) & 1;
#pragma unroll
    for (int nt = 0; nt < COUT / 16; ++nt) {
        int col = nt * 16 + l;
        float bb = bias[col];
        float s1 = 0.f, s2 = 0.f;
        size_t grow = (size_t)rowbase + wv * 16 + hi * 8;
#pragma unroll
        for (int j = 0; j < 8; ++j) {
            float v = acc[nt][j] + bb;
            s1 += v; s2 += v * v;
            yOut[(grow + j) * COUT + col] = (_Float16)v;
        }
        s1 += __shfl_xor(s1, 16);
        s2 += __shfl_xor(s2, 16);
        if (hi == 0) { atomicAdd(&ldsSum[col], s1); atomicAdd(&ldsSq[col], s2); }
    }
}

// ---------- One-shot: convert weights to f16, rows pre-padded to KP strides ----------
__global__ __launch_bounds__(256) void prep_weights(
    const float* __restrict__ w0, const float* __restrict__ w1, const float* __restrict__ w2,
    _Float16* __restrict__ p0, _Float16* __restrict__ p1, _Float16* __restrict__ p2)
{
    int t = blockIdx.x * 256 + threadIdx.x;
    int stride = gridDim.x * 256;
    for (int i = t; i < 64 * KP0; i += stride) {
        int o = i / KP0, c = i - o * KP0;
        p0[i] = (c < 73) ? (_Float16)w0[o * 73 + c] : (_Float16)0.f;
    }
    for (int i = t; i < 64 * KP1; i += stride) {
        int o = i / KP1, c = i - o * KP1;
        p1[i] = (c < 64) ? (_Float16)w1[(o << 6) + c] : (_Float16)0.f;
    }
    for (int i = t; i < 128 * KP1; i += stride) {
        int o = i / KP1, c = i - o * KP1;
        p2[i] = (c < 64) ? (_Float16)w2[(o << 6) + c] : (_Float16)0.f;
    }
}

// ---------- Layer 0: gather + geometry + GEMM 73->64 + stats ----------
__global__ __launch_bounds__(256) void gemm0_kernel(
    const float* __restrict__ center, const float* __restrict__ normal,
    const float* __restrict__ feature, const int* __restrict__ group_idx,
    const _Float16* __restrict__ prew, const float* __restrict__ b0,
    _Float16* __restrict__ y0, float* __restrict__ statsOut)
{
    __shared__ __align__(16) _Float16 As[ROWS * KP0];
    __shared__ __align__(16) _Float16 Ws[64 * KP0];
    __shared__ float ldsSum[64], ldsSq[64];
    int tid = threadIdx.x;
    int rowbase = blockIdx.x * ROWS;

    if (tid < 32) {                       // wave 0: DMA the weight tile to LDS
        tdm_load_1d_f16(Ws, prew, 64 * KP0);
        __builtin_amdgcn_s_wait_tensorcnt(0);
    }
    if (tid < 64) { ldsSum[tid] = 0.f; ldsSq[tid] = 0.f; }

    // Build the 128x73 input tile: [rel xyz | rho theta phi | normal | 64 feat]
    if (tid < ROWS) {
        int g = rowbase + tid;            // flat (point, sample)
        int p = g >> 5;
        int idx = group_idx[g];
        float cx = center[3*p],   cy = center[3*p+1],   cz = center[3*p+2];
        float dx = center[3*idx] - cx, dy = center[3*idx+1] - cy, dz = center[3*idx+2] - cz;
        float rho = sqrtf(dx*dx + dy*dy + dz*dz);
        float th = 0.f;
        if (rho > 0.f) th = acosf(fminf(fmaxf(dz / rho, -1.f), 1.f)) * 0.3183098861837907f;
        float ph = atan2f(dy, dx) * 0.15915494309189535f + 0.5f;
        _Float16* row = As + tid * KP0;
        row[0] = (_Float16)dx;  row[1] = (_Float16)dy;  row[2] = (_Float16)dz;
        row[3] = (_Float16)rho; row[4] = (_Float16)th;  row[5] = (_Float16)ph;
        row[6] = (_Float16)normal[3*idx];
        row[7] = (_Float16)normal[3*idx+1];
        row[8] = (_Float16)normal[3*idx+2];
        const float4* f4 = (const float4*)(feature + (size_t)idx * 64);
#pragma unroll
        for (int q = 0; q < 16; ++q) {
            float4 v = f4[q];
            row[ 9 + 4*q] = (_Float16)v.x;
            row[10 + 4*q] = (_Float16)v.y;
            row[11 + 4*q] = (_Float16)v.z;
            row[12 + 4*q] = (_Float16)v.w;
        }
#pragma unroll
        for (int c = 73; c < 96; ++c) row[c] = (_Float16)0.f;
    }
    __syncthreads();

    int wv = tid >> 5, lane = tid & 31;
    v8f acc[4] = {};
#pragma unroll
    for (int kb = 0; kb < 96; kb += 32) {
        v16h a = frag_a(As, KP0, wv * 16, kb, lane);
#pragma unroll
        for (int nt = 0; nt < 4; ++nt) {
            v16h b = frag_b(Ws, KP0, nt * 16, kb, lane);
            acc[nt] = __builtin_amdgcn_wmma_f32_16x16x32_f16(
                false, a, false, b, (short)0, acc[nt], false, false);
        }
    }
    epilogue<64>(acc, b0, y0, ldsSum, ldsSq, rowbase, wv, lane);
    __syncthreads();
    if (tid < 64) {
        atomicAdd(&statsOut[tid],       ldsSum[tid]);
        atomicAdd(&statsOut[128 + tid], ldsSq[tid]);
    }
}

// ---------- Layers 1 & 2: BN+ReLU on the fly, GEMM 64->COUT + stats ----------
template<int COUT>
__global__ __launch_bounds__(256) void gemm_bn_kernel(
    const _Float16* __restrict__ yIn, const float* __restrict__ statsPrev,
    const _Float16* __restrict__ prew, const float* __restrict__ bias,
    _Float16* __restrict__ yOut, float* __restrict__ statsOut)
{
    __shared__ __align__(16) _Float16 As[ROWS * KP1];
    __shared__ __align__(16) _Float16 Ws[COUT * KP1];
    __shared__ float scA[64], shA[64];
    __shared__ float ldsSum[COUT], ldsSq[COUT];
    int tid = threadIdx.x;
    int rowbase = blockIdx.x * ROWS;

    if (tid < 32) {                       // wave 0: DMA the weight tile to LDS
        tdm_load_1d_f16(Ws, prew, COUT * KP1);
        __builtin_amdgcn_s_wait_tensorcnt(0);
    }
    if (tid < 64) { scA[tid] = statsPrev[256 + tid]; shA[tid] = statsPrev[384 + tid]; }
    if (tid < COUT) { ldsSum[tid] = 0.f; ldsSq[tid] = 0.f; }
    __syncthreads();

    // Stage A: y_prev -> BN scale/shift -> ReLU -> f16 LDS. Half row per thread.
    {
        int r = tid >> 1, h = tid & 1;
        const _Float16* src = yIn + ((size_t)rowbase + r) * 64 + h * 32;
        _Float16* dst = As + r * KP1 + h * 32;
#pragma unroll
        for (int q = 0; q < 4; ++q) {
            v8h v = *(const v8h*)(src + q * 8);
            v8h o;
#pragma unroll
            for (int j = 0; j < 8; ++j) {
                int c = h * 32 + q * 8 + j;
                float f = (float)v[j] * scA[c] + shA[c];
                o[j] = (_Float16)fmaxf(f, 0.f);
            }
            *(v8h*)(dst + q * 8) = o;
        }
    }
    __syncthreads();

    int wv = tid >> 5, lane = tid & 31;
    v8f acc[COUT / 16] = {};
#pragma unroll
    for (int kb = 0; kb < 64; kb += 32) {
        v16h a = frag_a(As, KP1, wv * 16, kb, lane);
#pragma unroll
        for (int nt = 0; nt < COUT / 16; ++nt) {
            v16h b = frag_b(Ws, KP1, nt * 16, kb, lane);
            acc[nt] = __builtin_amdgcn_wmma_f32_16x16x32_f16(
                false, a, false, b, (short)0, acc[nt], false, false);
        }
    }
    epilogue<COUT>(acc, bias, yOut, ldsSum, ldsSq, rowbase, wv, lane);
    __syncthreads();
    if (tid < COUT) {
        atomicAdd(&statsOut[tid],       ldsSum[tid]);
        atomicAdd(&statsOut[128 + tid], ldsSq[tid]);
    }
}

// ---------- Fold sums into BN scale/shift ----------
__global__ void finalize_kernel(float* __restrict__ stats, const float* __restrict__ g,
                                const float* __restrict__ be, int C, float invM)
{
    int c = threadIdx.x;
    if (c < C) {
        float mean = stats[c] * invM;
        float var  = fmaxf(stats[128 + c] * invM - mean * mean, 0.f);
        float sc   = g[c] * rsqrtf(var + 1e-5f);
        stats[256 + c] = sc;
        stats[384 + c] = be[c] - mean * sc;
    }
}

// ---------- BN2 + ReLU + max-pool over 32 neighbors (8 channels/thread) ----------
__global__ __launch_bounds__(256) void pool_kernel(const _Float16* __restrict__ y2,
                                                   const float* __restrict__ stats2,
                                                   float* __restrict__ outFeat)
{
    int T = blockIdx.x * 256 + threadIdx.x;     // NPTS*16 threads
    int p = T >> 4, c0 = (T & 15) * 8;
    float sc[8], sh[8], m[8];
#pragma unroll
    for (int j = 0; j < 8; ++j) {
        sc[j] = stats2[256 + c0 + j];
        sh[j] = stats2[384 + c0 + j];
        m[j] = 0.f;                              // relu(max) == max(relu)
    }
    const _Float16* base = y2 + ((size_t)p * NSAMP) * 128 + c0;
    __builtin_prefetch(base + (size_t)16 * NSAMP * 128, 0, 1);
#pragma unroll 4
    for (int s = 0; s < NSAMP; ++s) {
        v8h v = *(const v8h*)(base + (size_t)s * 128);
#pragma unroll
        for (int j = 0; j < 8; ++j)
            m[j] = fmaxf(m[j], (float)v[j] * sc[j] + sh[j]);
    }
    float4* o4 = (float4*)(outFeat + (size_t)p * 128 + c0);
    o4[0] = make_float4(m[0], m[1], m[2], m[3]);
    o4[1] = make_float4(m[4], m[5], m[6], m[7]);
}

// ---------- Pass-through outputs: center, normal, offset ----------
__global__ void copy_misc(const float* __restrict__ center, const float* __restrict__ normal,
                          const int* __restrict__ offset, float* __restrict__ out)
{
    int i = blockIdx.x * blockDim.x + threadIdx.x;
    if (i < NPTS * 3)            out[i] = center[i];
    else if (i < NPTS * 6)       out[i] = normal[i - NPTS * 3];
    if (i == 0) ((int*)out)[NPTS * 6 + (size_t)NPTS * 128] = offset[0];
}

extern "C" void kernel_launch(void* const* d_in, const int* in_sizes, int n_in,
                              void* d_out, int out_size, void* d_ws, size_t ws_size,
                              hipStream_t stream) {
    const float* center    = (const float*)d_in[0];
    const float* normal    = (const float*)d_in[1];
    const float* feature   = (const float*)d_in[2];
    const int*   offset    = (const int*)  d_in[3];
    const int*   group_idx = (const int*)  d_in[4];
    const float* w0 = (const float*)d_in[5];
    const float* b0 = (const float*)d_in[6];
    const float* g0 = (const float*)d_in[7];
    const float* be0= (const float*)d_in[8];
    const float* w1 = (const float*)d_in[9];
    const float* b1 = (const float*)d_in[10];
    const float* g1 = (const float*)d_in[11];
    const float* be1= (const float*)d_in[12];
    const float* w2 = (const float*)d_in[13];
    const float* b2 = (const float*)d_in[14];
    const float* g2 = (const float*)d_in[15];
    const float* be2= (const float*)d_in[16];
    float* out = (float*)d_out;

    char* ws = (char*)d_ws;
    _Float16* y0 = (_Float16*)(ws);                                   // M*64 f16
    _Float16* y1 = (_Float16*)(ws + (size_t)MROWS * 64 * 2);          // M*64 f16
    _Float16* y2 = (_Float16*)(ws + (size_t)MROWS * 64 * 4);          // M*128 f16
    size_t statsOff = (size_t)MROWS * 64 * 4 + (size_t)MROWS * 128 * 2;
    float* stats = (float*)(ws + statsOff);       // 3 x 512 floats: sum|sumsq|scale|shift
    _Float16* p0 = (_Float16*)(ws + statsOff + 8192);                 // 64x104 f16
    _Float16* p1 = (_Float16*)(ws + statsOff + 8192 + 16384);         // 64x72 f16
    _Float16* p2 = (_Float16*)(ws + statsOff + 8192 + 32768);         // 128x72 f16

    const float invM = 1.0f / (float)MROWS;
    const int nblk = MROWS / ROWS;   // 8192

    hipMemsetAsync(stats, 0, 3 * 512 * sizeof(float), stream);
    prep_weights<<<20, 256, 0, stream>>>(w0, w1, w2, p0, p1, p2);
    copy_misc<<<(NPTS * 6 + 255) / 256, 256, 0, stream>>>(center, normal, offset, out);

    gemm0_kernel<<<nblk, 256, 0, stream>>>(center, normal, feature, group_idx,
                                           p0, b0, y0, stats);
    finalize_kernel<<<1, 128, 0, stream>>>(stats, g0, be0, 64, invM);

    gemm_bn_kernel<64><<<nblk, 256, 0, stream>>>(y0, stats, p1, b1, y1, stats + 512);
    finalize_kernel<<<1, 128, 0, stream>>>(stats + 512, g1, be1, 64, invM);

    gemm_bn_kernel<128><<<nblk, 256, 0, stream>>>(y1, stats + 512, p2, b2, y2, stats + 1024);
    finalize_kernel<<<1, 128, 0, stream>>>(stats + 1024, g2, be2, 128, invM);

    pool_kernel<<<(NPTS * 16) / 256, 256, 0, stream>>>(y2, stats + 1024, out + NPTS * 6);
}
